// PhysBiformerBlock_40853728920304
// MI455X (gfx1250) — matrix operand order repaired and checked by reading, not compiled, not run on hardware
//
#include <hip/hip_runtime.h>
#include <hip/hip_bf16.h>
#include <math.h>

typedef __attribute__((ext_vector_type(16))) _Float16 v16h;
typedef __attribute__((ext_vector_type(8)))  _Float16 v8h;
typedef __attribute__((ext_vector_type(8)))  float    v8f;

#define T_STEPS   4
#define BATCH     2
#define LTOT      4096
#define DIM       256
#define ROWS      32768      /* T*B*LTOT */
#define PER_T     2097152    /* B*LTOT*DIM */
#define NREG      64
#define REGION_SZ 64
#define NHEADS    8
#define HDIM      32
#define NKEY      256        /* TOPK * REGION */

__device__ __forceinline__ v16h make16(v8h lo, v8h hi) {
  return __builtin_shufflevector(lo, hi, 0,1,2,3,4,5,6,7,8,9,10,11,12,13,14,15);
}
// A-matrix lane load: 16 contiguous K values per half-wave
__device__ __forceinline__ v16h ldA16(const _Float16* p) {
  return make16(*(const v8h*)p, *(const v8h*)(p + 8));
}
// B-matrix lane load: 16 contiguous K values per half-wave
__device__ __forceinline__ v16h ldB16(const _Float16* p) {
  return make16(*(const v8h*)p, *(const v8h*)(p + 8));
}
__device__ __forceinline__ v8f wmma_f16(v16h a, v16h b, v8f c) {
  return __builtin_amdgcn_wmma_f32_16x16x32_f16(false, a, false, b, (short)0, c, false, false);
}

// CDNA5 async copy global->LDS, 16B per lane (tracked by ASYNCcnt)
__device__ __forceinline__ void async_ld128(unsigned lds_off, unsigned gbyte_off, const void* base) {
  asm volatile("global_load_async_to_lds_b128 %0, %1, %2"
               :: "v"(lds_off), "v"(gbyte_off), "s"(base) : "memory");
}
__device__ __forceinline__ void wait_async3() {
  asm volatile("s_wait_asynccnt 0x3" ::: "memory");
}
__device__ __forceinline__ void wait_async0() {
  asm volatile("s_wait_asynccnt 0x0" ::: "memory");
}

// ---------------- weight f32 -> f16 ----------------
__global__ void cvt_f16_kernel(const float* __restrict__ src, _Float16* __restrict__ dst, int n) {
  int i = blockIdx.x * 256 + threadIdx.x;
  if (i < n) dst[i] = (_Float16)src[i];
}

// ---------------- BN stats (deterministic two-pass) ----------------
__global__ void bn_stats_kernel(const float* __restrict__ x, float* __restrict__ part) {
  int d = threadIdx.x;                       // channel
  size_t r0 = (size_t)blockIdx.x * 128;      // 128 rows per block, 256 blocks
  float s = 0.f, q = 0.f;
  for (int r = 0; r < 128; ++r) {
    float v = x[(r0 + r) * DIM + d];
    s += v; q += v * v;
  }
  part[(size_t)blockIdx.x * 512 + d]       = s;
  part[(size_t)blockIdx.x * 512 + 256 + d] = q;
}

__global__ void bn_finalize_kernel(const float* __restrict__ part,
                                   const float* __restrict__ g, const float* __restrict__ b,
                                   float* __restrict__ mr) {
  int d = threadIdx.x;
  float s = 0.f, q = 0.f;
  for (int i = 0; i < 256; ++i) { s += part[(size_t)i * 512 + d]; q += part[(size_t)i * 512 + 256 + d]; }
  float mean = s * (1.0f / 32768.0f);
  float var  = q * (1.0f / 32768.0f) - mean * mean;
  float sc = rsqrtf(var + 1e-5f) * g[d];
  mr[d] = mean; mr[256 + d] = sc; mr[512 + d] = b[d] - mean * sc;
}

// ---------------- fused BN apply + LIF scan over T, emit f16 spikes ----------------
__global__ void bn_lif_kernel(const float* __restrict__ x, const float* __restrict__ mr,
                              _Float16* __restrict__ spk) {
  size_t i = (size_t)blockIdx.x * 256 + threadIdx.x;   // over B*LTOT*DIM
  if (i >= (size_t)PER_T) return;
  int d = (int)(i & 255);
  float sc = mr[256 + d], sh = mr[512 + d];
  float v = 0.f;
  for (int t = 0; t < T_STEPS; ++t) {
    float xn = x[(size_t)t * PER_T + i] * sc + sh;
    v = 0.5f * (v + xn);                 // v += (x - v)/tau, tau=2
    float s = (v >= 1.0f) ? 1.0f : 0.0f;
    spk[(size_t)t * PER_T + i] = (_Float16)s;
    v *= (1.0f - s);                     // hard reset
  }
}

// ---------------- blocked WMMA GEMM with async-LDS double buffering ----------------
// C = A[M,K](f16) x W[N,K](f16)^T + bias
// macro-tile 128M x 64N per block (8 waves, each 32x32 = 2x2 WMMA tiles)
// EPI 0: outH = val (f16)   EPI 1: outH = gelu(val) (f16)   EPI 2: outF = resid + val*scale (f32)
template<int EPI>
__global__ void __launch_bounds__(256)
gemm_wmma_kernel(const _Float16* __restrict__ A, int lda,
                 const _Float16* __restrict__ W,
                 const float* __restrict__ bias,
                 int M, int N, int K,
                 _Float16* __restrict__ outH,
                 float* __restrict__ outF,
                 const float* __restrict__ resid,
                 const float* __restrict__ scalep) {
  __shared__ __align__(16) _Float16 As[2][128 * 32];   // 8KB per buffer
  __shared__ __align__(16) _Float16 Bs[2][64 * 32];    // 4KB per buffer

  const int tid  = threadIdx.x;
  const int lane = tid & 31;
  const int wid  = tid >> 5;
  const int nbN  = N >> 6;
  const int mb   = blockIdx.x / nbN;
  const int nb   = blockIdx.x - mb * nbN;
  const unsigned m0 = (unsigned)mb * 128u;
  const unsigned n0 = (unsigned)nb * 64u;

  // staging mapping: thread t moves 16B: panel row t/4, col-halfs (t%4)*8
  const unsigned srow = (unsigned)tid >> 2;
  const unsigned scol = ((unsigned)tid & 3u) * 8u;
  const unsigned ldsA0 = (unsigned)(size_t)&As[0][0];
  const unsigned ldsB0 = (unsigned)(size_t)&Bs[0][0];
  const unsigned lA_t = (srow * 32u + scol) * 2u;

  auto prefetch = [&](int buf, unsigned kk) {
    unsigned a_off1 = ((m0 + srow) * (unsigned)lda + kk + scol) * 2u;
    unsigned a_off2 = ((m0 + 64u + srow) * (unsigned)lda + kk + scol) * 2u;
    unsigned b_off  = ((n0 + srow) * (unsigned)K + kk + scol) * 2u;
    unsigned lA = ldsA0 + (unsigned)buf * (128u * 32u * 2u) + lA_t;
    unsigned lB = ldsB0 + (unsigned)buf * (64u * 32u * 2u) + lA_t;
    async_ld128(lA, a_off1, A);
    async_ld128(lA + 64u * 32u * 2u, a_off2, A);
    async_ld128(lB, b_off, W);
  };

  const int wm = wid & 3;          // 0..3 : M sub-block
  const int wn = wid >> 2;         // 0..1 : N sub-block
  const int r     = lane & 15;
  const int koffA = (lane >> 4) << 4;
  const int koffB = (lane >> 4) << 4;

  v8f acc00 = {}, acc01 = {}, acc10 = {}, acc11 = {};

  prefetch(0, 0);
  const int nk = K >> 5;
  for (int kt = 0; kt < nk; ++kt) {
    if (kt + 1 < nk) { prefetch((kt + 1) & 1, (unsigned)(kt + 1) * 32u); wait_async3(); }
    else             { wait_async0(); }
    __syncthreads();                       // buffer kt&1 landed for all waves
    const _Float16* as = &As[kt & 1][0];
    const _Float16* bs = &Bs[kt & 1][0];
    v16h a0 = ldA16(as + (wm * 32 +  0 + r) * 32 + koffA);
    v16h a1 = ldA16(as + (wm * 32 + 16 + r) * 32 + koffA);
    v16h b0 = ldB16(bs + (wn * 32 +  0 + r) * 32 + koffB);
    v16h b1 = ldB16(bs + (wn * 32 + 16 + r) * 32 + koffB);
    acc00 = wmma_f16(a0, b0, acc00);
    acc01 = wmma_f16(a0, b1, acc01);
    acc10 = wmma_f16(a1, b0, acc10);
    acc11 = wmma_f16(a1, b1, acc11);
    __syncthreads();                       // reads done before next prefetch overwrites
  }

  const int mlane = (lane >> 4) << 3;
  const float sc = scalep ? scalep[0] : 0.0f;
#pragma unroll
  for (int ti = 0; ti < 2; ++ti) {
#pragma unroll
    for (int tj = 0; tj < 2; ++tj) {
      const v8f* ac = (ti == 0) ? (tj == 0 ? &acc00 : &acc01)
                                : (tj == 0 ? &acc10 : &acc11);
      const unsigned n = n0 + (unsigned)(wn * 32 + tj * 16 + r);
      const float bsn = bias ? bias[n] : 0.0f;
#pragma unroll
      for (int v = 0; v < 8; ++v) {
        const size_t m = (size_t)m0 + (unsigned)(wm * 32 + ti * 16 + mlane + v);
        float val = (*ac)[v] + bsn;
        if (EPI == 0) {
          outH[m * N + n] = (_Float16)val;
        } else if (EPI == 1) {
          outH[m * N + n] = (_Float16)(0.5f * val * (1.0f + erff(val * 0.70710678118f)));
        } else {
          outF[m * N + n] = resid[m * N + n] + val * sc;
        }
      }
    }
  }
}

// ---------------- region mean pool of q and k ----------------
__global__ void region_mean_kernel(const _Float16* __restrict__ qkv,
                                   float* __restrict__ qm, float* __restrict__ km) {
  int blk = blockIdx.x;      // tb*64 + reg, 512 blocks
  int d = threadIdx.x;       // 256
  size_t row0 = (size_t)blk * REGION_SZ;
  float sq = 0.f, sk = 0.f;
  for (int r = 0; r < REGION_SZ; ++r) {
    const _Float16* p = qkv + (row0 + r) * 768;
    sq += (float)p[d];
    sk += (float)p[256 + d];
  }
  qm[(size_t)blk * 256 + d] = sq * (1.0f / 64.0f);
  km[(size_t)blk * 256 + d] = sk * (1.0f / 64.0f);
}

// ---------------- routing: affinity + top-4 (descending, first-index ties) ----------------
__global__ void route_topk_kernel(const float* __restrict__ qm, const float* __restrict__ km,
                                  int* __restrict__ ridx) {
  __shared__ float aff[NREG];
  int tb = blockIdx.x >> 6, n = blockIdx.x & 63;
  int m = threadIdx.x;       // 64 threads
  const float* qv = qm + ((size_t)tb * NREG + n) * 256;
  const float* kv = km + ((size_t)tb * NREG + m) * 256;
  float dot = 0.f;
  for (int d = 0; d < 256; ++d) dot += qv[d] * kv[d];
  aff[m] = dot;
  __syncthreads();
  if (m == 0) {
    int chosen[4];
    for (int t = 0; t < 4; ++t) {
      float best = -3.4e38f; int bi = 0;
      for (int j = 0; j < NREG; ++j) {
        bool taken = false;
        for (int u = 0; u < t; ++u) taken = taken || (chosen[u] == j);
        if (!taken && aff[j] > best) { best = aff[j]; bi = j; }
      }
      chosen[t] = bi;
      ridx[(size_t)blockIdx.x * 4 + t] = bi;
    }
  }
}

// ---------------- attention: per (tb, region, head); 4 waves x 16 q-rows ----------------
__global__ void __launch_bounds__(128)
attn_wmma_kernel(const _Float16* __restrict__ qkv, const int* __restrict__ ridx,
                 _Float16* __restrict__ o) {
  extern __shared__ char smem[];
  float*    scores = (float*)smem;                         // [64][256] f32, reused as f16 probs
  _Float16* vt     = (_Float16*)(smem + 64 * 256 * 4);     // [32][256] transposed gathered V

  const int blk  = blockIdx.x;
  const int head = blk & 7;
  const int reg  = (blk >> 3) & 63;
  const int tb   = blk >> 9;
  const size_t rowbase = (size_t)tb * LTOT;
  const int lane = threadIdx.x & 31;
  const int wid  = threadIdx.x >> 5;
  const int* rix = ridx + ((size_t)tb * NREG + reg) * 4;

  // gather V for this head, transposed: vt[d][key]
  for (int e = threadIdx.x; e < NKEY * HDIM; e += 128) {
    int key = e >> 5, d = e & 31;
    int token = rix[key >> 6] * REGION_SZ + (key & 63);
    vt[d * NKEY + key] = qkv[(rowbase + token) * 768 + 512 + head * 32 + d];
  }
  __syncthreads();

  const int qbase = wid * 16;
  const int r     = lane & 15;
  const int koffA = (lane >> 4) << 4;
  const int koffB = (lane >> 4) << 4;

  // Q A-tile: rows qbase..qbase+15, K = 32 head dims
  v16h aq = ldA16(qkv + (rowbase + reg * REGION_SZ + qbase + r) * 768 + head * 32 + koffA);

  // logits = Q x Kg^T, 16 key-tiles
  for (int kt = 0; kt < 16; ++kt) {
    int key = kt * 16 + r;
    int token = rix[key >> 6] * REGION_SZ + (key & 63);
    v16h bk = ldB16(qkv + (rowbase + token) * 768 + 256 + head * 32 + koffB);
    v8f c = {};
    c = wmma_f16(aq, bk, c);
    int nn = kt * 16 + r;
    int m0 = qbase + ((lane >> 4) << 3);
#pragma unroll
    for (int v = 0; v < 8; ++v) scores[(m0 + v) * NKEY + nn] = c[v];
  }
  __syncthreads();

  // row softmax (scale = hd^-0.5), convert in-place to f16 probs at row base
  const float sm = 0.17677669529663687f;
  if (lane < 16) {
    float* srow = scores + (qbase + lane) * NKEY;
    float mx = -3.4e38f;
    for (int j = 0; j < NKEY; ++j) mx = fmaxf(mx, srow[j] * sm);
    float sum = 0.f;
    for (int j = 0; j < NKEY; ++j) sum += expf(srow[j] * sm - mx);
    float inv = 1.0f / sum;
    _Float16* prow = (_Float16*)srow;
    for (int j = 0; j < NKEY; ++j) {
      float p = expf(srow[j] * sm - mx) * inv;   // reads f32[j] before clobbering bytes [2j,2j+2)
      prow[j] = (_Float16)p;
    }
  }
  __syncthreads();

  // O = P x Vg : K loop over 256 keys, two 16-wide N tiles covering hd=32
  v8f oc0 = {}, oc1 = {};
  const _Float16* prow = (const _Float16*)(scores + (qbase + r) * NKEY);
  for (int kc = 0; kc < 8; ++kc) {
    v16h ap = ldA16(prow + kc * 32 + koffA);
    v16h b0 = ldB16(vt + (0 * 16 + r) * NKEY + kc * 32 + koffB);
    v16h b1 = ldB16(vt + (1 * 16 + r) * NKEY + kc * 32 + koffB);
    oc0 = wmma_f16(ap, b0, oc0);
    oc1 = wmma_f16(ap, b1, oc1);
  }
  int m0 = qbase + ((lane >> 4) << 3);
#pragma unroll
  for (int v = 0; v < 8; ++v) {
    size_t orow = (rowbase + reg * REGION_SZ + m0 + v) * DIM;
    o[orow + head * 32 + r]      = (_Float16)oc0[v];
    o[orow + head * 32 + 16 + r] = (_Float16)oc1[v];
  }
}

// =======================================================================
extern "C" void kernel_launch(void* const* d_in, const int* in_sizes, int n_in,
                              void* d_out, int out_size, void* d_ws, size_t ws_size,
                              hipStream_t stream) {
  (void)in_sizes; (void)n_in; (void)out_size; (void)ws_size;
  const float* x      = (const float*)d_in[0];
  const float* bn1_g  = (const float*)d_in[1];
  const float* bn1_b  = (const float*)d_in[2];
  const float* bn2_g  = (const float*)d_in[3];
  const float* bn2_b  = (const float*)d_in[4];
  const float* qkv_w  = (const float*)d_in[5];
  const float* qkv_b  = (const float*)d_in[6];
  const float* proj_w = (const float*)d_in[7];
  const float* proj_b = (const float*)d_in[8];
  const float* ffn_w1 = (const float*)d_in[9];
  const float* ffn_b1 = (const float*)d_in[10];
  const float* ffn_w2 = (const float*)d_in[11];
  const float* ffn_b2 = (const float*)d_in[12];
  const float* scale  = (const float*)d_in[13];
  float* out = (float*)d_out;

  char* ws = (char*)d_ws;
  size_t off = 0;
  auto carve = [&](size_t bytes) -> void* {
    void* p = ws + off; off += (bytes + 255) & ~(size_t)255; return p;
  };
  float*     part   = (float*)carve(256 * 512 * sizeof(float));
  float*     mr     = (float*)carve(768 * sizeof(float));
  _Float16*  w_qkv  = (_Float16*)carve((size_t)768 * 256 * 2);
  _Float16*  w_proj = (_Float16*)carve((size_t)256 * 256 * 2);
  _Float16*  w_f1   = (_Float16*)carve((size_t)1024 * 256 * 2);
  _Float16*  w_f2   = (_Float16*)carve((size_t)256 * 1024 * 2);
  _Float16*  spk    = (_Float16*)carve((size_t)ROWS * DIM * 2);
  _Float16*  qkvb   = (_Float16*)carve((size_t)ROWS * 768 * 2);
  float*     qm     = (float*)carve((size_t)512 * 256 * 4);
  float*     km     = (float*)carve((size_t)512 * 256 * 4);
  int*       ridx   = (int*)carve((size_t)512 * 4 * 4);
  _Float16*  obuf   = (_Float16*)carve((size_t)ROWS * DIM * 2);
  float*     x2     = (float*)carve((size_t)ROWS * DIM * 4);
  _Float16*  hbuf   = (_Float16*)carve((size_t)ROWS * 1024 * 2);

  // weights -> f16
  cvt_f16_kernel<<<768, 256, 0, stream>>>(qkv_w, w_qkv, 768 * 256);
  cvt_f16_kernel<<<256, 256, 0, stream>>>(proj_w, w_proj, 256 * 256);
  cvt_f16_kernel<<<1024, 256, 0, stream>>>(ffn_w1, w_f1, 1024 * 256);
  cvt_f16_kernel<<<1024, 256, 0, stream>>>(ffn_w2, w_f2, 256 * 1024);

  // ---- BN1 + LIF -> spikes ----
  bn_stats_kernel<<<256, 256, 0, stream>>>(x, part);
  bn_finalize_kernel<<<1, 256, 0, stream>>>(part, bn1_g, bn1_b, mr);
  bn_lif_kernel<<<PER_T / 256, 256, 0, stream>>>(x, mr, spk);

  // ---- QKV GEMM: [32768,256]x[768,256]^T -> f16 [32768,768] ----
  gemm_wmma_kernel<0><<<(ROWS / 128) * (768 / 64), 256, 0, stream>>>(
      spk, DIM, w_qkv, qkv_b, ROWS, 768, DIM, qkvb, nullptr, nullptr, nullptr);

  // ---- routing ----
  region_mean_kernel<<<512, 256, 0, stream>>>(qkvb, qm, km);
  route_topk_kernel<<<512, 64, 0, stream>>>(qm, km, ridx);

  // ---- bilevel attention ----
  attn_wmma_kernel<<<8 * NREG * NHEADS, 128, 64 * 256 * 4 + 32 * 256 * 2, stream>>>(
      qkvb, ridx, obuf);

  // ---- proj + residual: x2 = x + (o@Wp^T + b)*scale ----
  gemm_wmma_kernel<2><<<(ROWS / 128) * (256 / 64), 256, 0, stream>>>(
      obuf, DIM, w_proj, proj_b, ROWS, DIM, DIM, nullptr, x2, x, scale);

  // ---- BN2 + LIF -> spikes (reuse spk) ----
  bn_stats_kernel<<<256, 256, 0, stream>>>(x2, part);
  bn_finalize_kernel<<<1, 256, 0, stream>>>(part, bn2_g, bn2_b, mr);
  bn_lif_kernel<<<PER_T / 256, 256, 0, stream>>>(x2, mr, spk);

  // ---- FFN1 + exact GELU -> f16 [32768,1024] ----
  gemm_wmma_kernel<1><<<(ROWS / 128) * (1024 / 64), 256, 0, stream>>>(
      spk, DIM, w_f1, ffn_b1, ROWS, 1024, DIM, hbuf, nullptr, nullptr, nullptr);

  // ---- FFN2 + residual -> out = x2 + (h@W2^T + b2)*scale ----
  gemm_wmma_kernel<2><<<(ROWS / 128) * (256 / 64), 256, 0, stream>>>(
      hbuf, 1024, w_f2, ffn_b2, ROWS, DIM, 1024, nullptr, out, x2, scale);
}
